// RealizedGARCHPINNv2_55087250538690
// MI455X (gfx1250) — matrix operand order repaired and checked by reading, not compiled, not run on hardware
//
#include <hip/hip_runtime.h>

#define B_TOT 2048
#define T_LEN 2048
#define HID   128
#define IPD   78

typedef _Float16 v16h __attribute__((ext_vector_type(16)));
typedef _Float16 v8h  __attribute__((ext_vector_type(8)));
typedef float    v8f  __attribute__((ext_vector_type(8)));

// Intra-wave LDS producer->consumer ordering (CDNA5 split counters).
#define LDS_FENCE() asm volatile("s_wait_dscnt 0" ::: "memory")

__device__ __forceinline__ v8h leaky_pk(v8h x) {
    // LeakyReLU(0.1) == max(x, 0.1x): v_pk_mul_f16 + v_pk_max_num_f16
    return __builtin_elementwise_max(x, x * (_Float16)0.1f);
}

__global__ __launch_bounds__(64, 1)
void garch_pinn_wmma_kernel(const float* __restrict__ returns,
                            const float* __restrict__ log_rv,
                            const float* __restrict__ p_omega, const float* __restrict__ p_beta,
                            const float* __restrict__ p_tau1,  const float* __restrict__ p_tau2,
                            const float* __restrict__ p_gamma, const float* __restrict__ p_xi,
                            const float* __restrict__ p_phi,   const float* __restrict__ p_d1,
                            const float* __restrict__ p_d2,    const float* __restrict__ p_mu,
                            const float* __restrict__ W1, const float* __restrict__ b1,
                            const float* __restrict__ W2, const float* __restrict__ b2,
                            const float* __restrict__ W3, const float* __restrict__ b3,
                            const float* __restrict__ W4, const float* __restrict__ b4,
                            float* __restrict__ out_lh, float* __restrict__ out_lx,
                            float* __restrict__ out_z,  float* __restrict__ out_u) {
    // Weights pre-swizzled as WMMA *A*-fragments (Wt = W transposed: out x in).
    // A-frag layout (16-bit A 16x32): lane m=l&15; halfs 0-15 -> K=g*16..g*16+15
    // (g=l>>4), matching the contiguous B-frag. Stored lane-major, 16 halfs contiguous.
    __shared__ __align__(32) _Float16 sW1[8  * 512];   // ktiles=1, mtiles=8 (K pad 5->32, row5=b1)
    __shared__ __align__(32) _Float16 sW2[32 * 512];   // ktiles=4, mtiles=8
    __shared__ __align__(32) _Float16 sW3[16 * 512];   // ktiles=4, mtiles=4
    __shared__ __align__(32) _Float16 sFeat[2 * 16 * 32];   // per-wave 16x32; col5=1, cols6..31=0
    __shared__ __align__(32) _Float16 sAct1[2 * 16 * HID];  // row-major [batch][chan]
    __shared__ __align__(32) _Float16 sAct2[2 * 16 * HID];
    __shared__ __align__(32) _Float16 sB2h[HID];

    const int tid = threadIdx.x;

    // ---- one-time weight conversion + A-fragment swizzle into LDS ----
    for (int p = tid; p < 8 * 512; p += 64) {        // Wt1: 128x32 (b1 folded at K=5)
        int j = p & 15, ln = (p >> 4) & 31, mt = p >> 9;
        int gg = ln >> 4;
        int k = gg * 16 + j;
        int m = mt * 16 + (ln & 15);
        _Float16 v = (_Float16)0.f;
        if (k < 5)       v = (_Float16)W1[k * 128 + m];
        else if (k == 5) v = (_Float16)b1[m];
        sW1[p] = v;
    }
    for (int p = tid; p < 32 * 512; p += 64) {       // Wt2: 128x128
        int j = p & 15, ln = (p >> 4) & 31, f = p >> 9;
        int mt = f & 7, kt = f >> 3;
        int gg = ln >> 4;
        int k = kt * 32 + gg * 16 + j;
        int m = mt * 16 + (ln & 15);
        sW2[p] = (_Float16)W2[k * 128 + m];
    }
    for (int p = tid; p < 16 * 512; p += 64) {       // Wt3: 64x128
        int j = p & 15, ln = (p >> 4) & 31, f = p >> 9;
        int mt = f & 3, kt = f >> 2;
        int gg = ln >> 4;
        int k = kt * 32 + gg * 16 + j;
        int m = mt * 16 + (ln & 15);
        sW3[p] = (_Float16)W3[k * 64 + m];
    }
    for (int i = tid; i < HID; i += 64) sB2h[i] = (_Float16)b2[i];
    for (int p = tid; p < 2 * 16 * 32; p += 64) sFeat[p] = (_Float16)0.f;
    __syncthreads();
    if (tid < 32) {                                  // constant-1 feature column (bias fold)
        sFeat[(tid & 15) * 32 + 5 + (tid >> 4) * 16 * 32] = (_Float16)1.f;
    }
    __syncthreads();

    const int lane = tid & 31;
    const int wave = tid >> 5;
    const int lm   = lane & 15;     // batch row within tile (and D-frag n)
    const int g    = lane >> 4;     // half-wave group
    const int row  = blockIdx.x * 32 + wave * 16 + lm;   // lanes 16-31 mirror 0-15

    const float omega = p_omega[0], beta = p_beta[0], tau1 = p_tau1[0], tau2 = p_tau2[0];
    const float gam = p_gamma[0], xi = p_xi[0], phi = p_phi[0];
    const float d1 = p_d1[0], d2 = p_d2[0], mu = p_mu[0];
    const float b4v = b4[0];

    // Register-resident b3/W4 slices for the fused layer3+layer4 epilogue:
    // lane holds channels ch = mt*16 + g*8 + i  (constant over all time steps).
    float b3r[32], w4r[32];
    #pragma unroll
    for (int mt = 0; mt < 4; ++mt)
        #pragma unroll
        for (int i = 0; i < 8; ++i) {
            int ch = mt * 16 + g * 8 + i;
            b3r[mt * 8 + i] = b3[ch];
            w4r[mt * 8 + i] = W4[ch];
        }

    _Float16* feat = sFeat + wave * 16 * 32;
    _Float16* act1 = sAct1 + wave * 16 * HID;
    _Float16* act2 = sAct2 + wave * 16 * HID;

    // D-frag: n = lm (batch), m = mt*16 + g*8 + i (channel) -> 8 contiguous halfs per lane.
    _Float16* st1 = act1 + lm * HID + g * 8;
    _Float16* st2 = act2 + lm * HID + g * 8;

    const float* ret_row = returns + (size_t)row * T_LEN;
    const float* lrv_row = log_rv  + (size_t)row * T_LEN;
    const float log_h0 = lrv_row[0];   // log(exp(x)) == x

    float prev_lh = 0.f, prev_z = 0.f, prev_lx = 0.f;
    int tod_i = 0, dow_i = 0;

    for (int t = 0; t < T_LEN; ++t) {
        // ---- scalar recurrence (fp32 VALU; all lanes, rows mirrored) ----
        float r_t   = ret_row[t];
        float lrv_t = lrv_row[t];
        float lh = (t == 0) ? log_h0
                 : omega + beta * prev_lh + tau1 * prev_z
                   + tau2 * (prev_z * prev_z - 1.f) + gam * prev_lx;
        float z  = (r_t - mu) * __expf(-0.5f * lh);    // 1/sqrt(exp(lh))
        float lx = xi + phi * lh + d1 * z + d2 * (z * z - 1.f);
        float u  = lrv_t - lx;
        float tod = (float)tod_i * (1.f / 77.f);
        float dow = (float)dow_i * 0.25f;
        if (++tod_i == IPD) { tod_i = 0; if (++dow_i == 5) dow_i = 0; }

        if (g == 0) {
            _Float16* fr = feat + lm * 32;
            fr[0] = (_Float16)lh;  fr[1] = (_Float16)z;  fr[2] = (_Float16)u;
            fr[3] = (_Float16)tod; fr[4] = (_Float16)dow;   // fr[5] == 1 (bias column)
        }
        LDS_FENCE();

        // B-frag (activations, 32x16): lanes 0-15 hold K=0-15 of col n=lm,
        // lanes 16-31 hold K=16-31 -> one contiguous 32B load per lane.
        // ---- layer 1: Wt1(128x32-pad) x featT(32x16), bias folded ----
        v16h bf0 = *(const v16h*)(feat + lm * 32 + g * 16);
        #pragma unroll
        for (int mt = 0; mt < 8; ++mt) {
            v16h af = *(const v16h*)(sW1 + mt * 512 + lane * 16);
            v8f acc = {};
            acc = __builtin_amdgcn_wmma_f32_16x16x32_f16(false, af, false, bf0,
                                                         (short)0, acc, false, false);
            v8h x = __builtin_convertvector(acc, v8h);
            *(v8h*)(st1 + mt * 16) = leaky_pk(x);
        }
        LDS_FENCE();

        // ---- layer 2: Wt2(128x128) x act1T(128x16) ----
        v16h b2f[4];
        #pragma unroll
        for (int kt = 0; kt < 4; ++kt)
            b2f[kt] = *(const v16h*)(act1 + lm * HID + kt * 32 + g * 16);
        #pragma unroll
        for (int mt = 0; mt < 8; ++mt) {
            v8f acc = {};
            #pragma unroll
            for (int kt = 0; kt < 4; ++kt) {
                v16h af = *(const v16h*)(sW2 + (kt * 8 + mt) * 512 + lane * 16);
                acc = __builtin_amdgcn_wmma_f32_16x16x32_f16(false, af, false, b2f[kt],
                                                             (short)0, acc, false, false);
            }
            v8h bh = *(const v8h*)(sB2h + mt * 16 + g * 8);
            v8h x = __builtin_convertvector(acc, v8h) + bh;
            *(v8h*)(st2 + mt * 16) = leaky_pk(x);
        }
        LDS_FENCE();

        // ---- layer 3 + layer 4 fused: Wt3(64x128) x act2T(128x16), then
        //      bias + leaky + dot(W4) in f32 directly on the accumulator; the
        //      two half-wave channel groups combine with one lane shuffle.
        v16h b3f[4];
        #pragma unroll
        for (int kt = 0; kt < 4; ++kt)
            b3f[kt] = *(const v16h*)(act2 + lm * HID + kt * 32 + g * 16);
        float s = 0.f;
        #pragma unroll
        for (int mt = 0; mt < 4; ++mt) {
            v8f acc = {};
            #pragma unroll
            for (int kt = 0; kt < 4; ++kt) {
                v16h af = *(const v16h*)(sW3 + (kt * 4 + mt) * 512 + lane * 16);
                acc = __builtin_amdgcn_wmma_f32_16x16x32_f16(false, af, false, b3f[kt],
                                                             (short)0, acc, false, false);
            }
            #pragma unroll
            for (int i = 0; i < 8; ++i) {
                float v = acc[i] + b3r[mt * 8 + i];
                v = fmaxf(v, 0.1f * v);              // LeakyReLU(0.1)
                s = fmaf(v, w4r[mt * 8 + i], s);
            }
        }
        s += __shfl_xor(s, 16, 32);                  // channels g=0 half + g=1 half
        float enh = lh + 0.01f * (s + b4v);

        if (g == 0) {
            size_t o = (size_t)row * T_LEN + t;
            out_lh[o] = enh;
            out_lx[o] = lx;
            out_z[o]  = z;
            out_u[o]  = u;
        }
        prev_lh = enh; prev_z = z; prev_lx = lx;
        // no loop-end fence: next iteration's post-feat-store s_wait_dscnt 0
        // covers the layer-3 B-frag reads before any act overwrite.
    }
}

extern "C" void kernel_launch(void* const* d_in, const int* in_sizes, int n_in,
                              void* d_out, int out_size, void* d_ws, size_t ws_size,
                              hipStream_t stream) {
    (void)in_sizes; (void)n_in; (void)d_ws; (void)ws_size; (void)out_size;
    const float* returns = (const float*)d_in[0];
    const float* log_rv  = (const float*)d_in[1];
    const float* omega   = (const float*)d_in[2];
    const float* beta    = (const float*)d_in[3];
    const float* tau1    = (const float*)d_in[4];
    const float* tau2    = (const float*)d_in[5];
    const float* gamma_  = (const float*)d_in[6];
    const float* xi      = (const float*)d_in[7];
    const float* phi     = (const float*)d_in[8];
    const float* delta1  = (const float*)d_in[9];
    const float* delta2  = (const float*)d_in[10];
    const float* mu      = (const float*)d_in[11];
    const float* W1 = (const float*)d_in[12]; const float* b1 = (const float*)d_in[13];
    const float* W2 = (const float*)d_in[14]; const float* b2 = (const float*)d_in[15];
    const float* W3 = (const float*)d_in[16]; const float* b3 = (const float*)d_in[17];
    const float* W4 = (const float*)d_in[18]; const float* b4 = (const float*)d_in[19];

    float* out = (float*)d_out;
    const size_t BT = (size_t)B_TOT * T_LEN;
    float* out_lh = out;
    float* out_lx = out + BT;
    float* out_z  = out + 2 * BT;
    float* out_u  = out + 3 * BT;

    dim3 grid(B_TOT / 32);   // 64 workgroups x 2 waves x 16 rows
    dim3 block(64);
    hipLaunchKernelGGL(garch_pinn_wmma_kernel, grid, block, 0, stream,
                       returns, log_rv, omega, beta, tau1, tau2, gamma_, xi, phi,
                       delta1, delta2, mu, W1, b1, W2, b2, W3, b3, W4, b4,
                       out_lh, out_lx, out_z, out_u);
}
